// Model_22608707846346
// MI455X (gfx1250) — matrix operand order repaired and checked by reading, not compiled, not run on hardware
//
#include <hip/hip_runtime.h>
#include <hip/hip_bf16.h>
#include <math.h>

typedef __attribute__((ext_vector_type(16))) __bf16 v16bf;
typedef __attribute__((ext_vector_type(8)))  float  v8f;

// Channel permutation (involution): swap bits 3 and 4. Feature rows are stored
// with each 32-channel block in order [0..7, 16..23, 8..15, 24..31], which makes
// every WMMA lane's 16 A-operand values one contiguous 32B load AND makes the
// hardware K slot equal the true cin (B weights are packed with the same chperm swizzle).
__host__ __device__ __forceinline__ int chperm(int c) {
    return (c & ~24) | ((c & 8) << 1) | ((c & 16) >> 1);
}

// ---------------------------------------------------------------------------
// Tiny helpers
// ---------------------------------------------------------------------------

__global__ void zero_bf16_kernel(__bf16* p, int n) {
    int t = blockIdx.x * blockDim.x + threadIdx.x;
    if (t < n) p[t] = (__bf16)0.0f;
}

// bn params p: rows gamma, beta, mean, var (each C). scale = g*rsqrt(var+eps),
// shift = beta - mean*scale.
__global__ void bn_prep_kernel(const float* __restrict__ p,
                               float* __restrict__ scale,
                               float* __restrict__ shift, int C) {
    int c = blockIdx.x * blockDim.x + threadIdx.x;
    if (c < C) {
        float s = p[c] * rsqrtf(p[3 * C + c] + 1e-4f);
        scale[c] = s;
        shift[c] = p[C + c] - p[2 * C + c] * s;
    }
}

// Pack f32 weights [K][Cin][Cout] -> bf16 [K][Cout][Cin] (cin stored at chperm(ci),
// the same channel swizzle as the feature buffers, so A and B K slots match).
__global__ void pack_w_kernel(const float* __restrict__ W, __bf16* __restrict__ Wt,
                              int K, int Cin, int Cout) {
    int t = blockIdx.x * blockDim.x + threadIdx.x;
    int total = K * Cin * Cout;
    if (t >= total) return;
    int k = t / (Cin * Cout);
    int r = t - k * (Cin * Cout);
    int ci = r / Cout;
    int co = r - ci * Cout;
    Wt[(k * Cout + co) * Cin + chperm(ci)] = (__bf16)W[t];
}

// ---------------------------------------------------------------------------
// Initial conv: K=125, Cin=1, Cout=32 (scalar, trivial FLOPs)
// ---------------------------------------------------------------------------
__global__ void conv0_kernel(const float* __restrict__ xf, const float* __restrict__ W,
                             const int* __restrict__ nbr, __bf16* __restrict__ Y, int N) {
    int t = blockIdx.x * blockDim.x + threadIdx.x;
    if (t >= N * 32) return;
    int j = t >> 5, co = t & 31;
    float acc = 0.0f;
    for (int k = 0; k < 125; ++k) {
        int idx = nbr[k * N + j];
        float xv = (idx < N) ? xf[idx] : 0.0f;
        acc += xv * W[k * 32 + co];
    }
    Y[j * 32 + chperm(co)] = (__bf16)acc;
}

// ---------------------------------------------------------------------------
// Core gather-GEMM with WMMA bf16 -> f32, templated on channel geometry.
// One wave = 16 output rows x ALL cout chunks (A gathered once, reused across
// CHUNKS accumulators -> CHUNKS-fold cut in gather bytes, the binding resource
// on this L2-resident workload). Buffers are row-padded to multiples of 16 so
// stores are unconditional (pad rows are never gathered).
// MODE: 1 = BN + leaky ; 2 = BN + residual add (resid/Y may alias in-place:
// each element is read+written only by its owning wave).
// ---------------------------------------------------------------------------
template <int CI, int CO, int MODE>
__global__ __launch_bounds__(256) void subm_conv_wmma(
    const __bf16* __restrict__ X, const __bf16* __restrict__ Wt,
    const int* __restrict__ nbr, const __bf16* __restrict__ zrow,
    const float* __restrict__ scale, const float* __restrict__ shift,
    const __bf16* __restrict__ resid, __bf16* __restrict__ Y,
    int K, int Nin, int Nout)
{
    constexpr int CHUNKS = CO / 16;   // cout chunks (accumulators)
    constexpr int CINCH  = CI / 32;   // 32-wide cin chunks

    const int lane = threadIdx.x & 31;
    const int wave = blockIdx.x * (blockDim.x >> 5) + (threadIdx.x >> 5);
    const int tiles = (Nout + 15) >> 4;
    if (wave >= tiles) return;        // wave-uniform: EXEC stays all-1s

    const int tile = wave;
    const int col  = lane & 15;
    const int hi   = lane >> 4;       // 0: lanes 0-15, 1: lanes 16-31

    // A: channel-swizzled rows -> lane's 16 values are contiguous at +hi*16.
    // B: lane holds column col with one contiguous 32B K half at +hi*16.
    const int jj0 = tile * 16 + col;
    const int jfetch = (jj0 < Nout) ? jj0 : (Nout - 1);

    v8f acc[CHUNKS] = {};             // zero-init all accumulators

    for (int k = 0; k < K; ++k) {
        const int idx = nbr[k * Nout + jfetch];
        const __bf16* xr = (idx < Nin) ? (X + (size_t)idx * CI) : zrow;
        const __bf16* wk = Wt + (size_t)k * CO * CI;
#pragma unroll
        for (int cb = 0; cb < CINCH; ++cb) {
            const v16bf a = *reinterpret_cast<const v16bf*>(xr + cb * 32 + hi * 16);
#pragma unroll
            for (int cc = 0; cc < CHUNKS; ++cc) {
                const v16bf b = *reinterpret_cast<const v16bf*>(
                    wk + (cc * 16 + col) * CI + cb * 32 + hi * 16);  // 32B contiguous
                acc[cc] = __builtin_amdgcn_wmma_f32_16x16x32_bf16(
                    false, a, false, b, (short)0, acc[cc], false, false);
            }
        }
    }

    const int jBase = tile * 16 + hi * 8;   // D row M = v + 8*hi
#pragma unroll
    for (int cc = 0; cc < CHUNKS; ++cc) {
        const int ch  = cc * 16 + col;      // true cout channel
        const int pos = chperm(ch);         // swizzled storage position
        const float s  = scale[ch];
        const float sh = shift[ch];
#pragma unroll
        for (int v = 0; v < 8; ++v) {
            const int j = jBase + v;        // unconditional: buffers row-padded
            float val = acc[cc][v] * s + sh;
            if (MODE == 1) {
                val = (val > 0.0f) ? val : 0.333f * val;
            } else {                        // MODE 2: residual add
                val += (float)resid[(size_t)j * CO + pos];
            }
            Y[(size_t)j * CO + pos] = (__bf16)val;
        }
    }
}

// ---------------------------------------------------------------------------
// Final conv: K=27, Cin=192, Cout=2 (scalar; N5 <= 512). Reads swizzled
// feature storage through chperm.
// ---------------------------------------------------------------------------
__global__ void fin_conv_kernel(const __bf16* __restrict__ X, const float* __restrict__ W,
                                const int* __restrict__ nbr, float* __restrict__ out, int N) {
    int t = blockIdx.x * blockDim.x + threadIdx.x;
    if (t >= N * 2) return;
    int j = t >> 1, co = t & 1;
    float acc = 0.0f;
    for (int k = 0; k < 27; ++k) {
        int idx = nbr[k * N + j];
        if (idx < N) {
            const __bf16* xr = X + (size_t)idx * 192;
            const float* wr = W + (size_t)k * 192 * 2 + co;
            for (int ci = 0; ci < 192; ++ci) acc += (float)xr[chperm(ci)] * wr[ci * 2];
        }
    }
    out[j * 2 + co] = acc;
}

// ---------------------------------------------------------------------------
// Head: dense scatter 8x8x8x2 -> flat[1024] (channel-major), linear->elu->
// linear->sigmoid. Single block.
// ---------------------------------------------------------------------------
__global__ void head_kernel(const float* __restrict__ fin, const int* __restrict__ c5, int N5,
                            const float* __restrict__ Wl, const float* __restrict__ bl,
                            const float* __restrict__ W3, const float* __restrict__ b3,
                            float* __restrict__ out) {
    __shared__ float flat[1024];
    __shared__ float part[256][2];
    const int t = threadIdx.x;
    for (int i = t; i < 1024; i += blockDim.x) flat[i] = 0.0f;
    __syncthreads();
    for (int e = t; e < N5 * 2; e += blockDim.x) {
        int j = e >> 1, c = e & 1;
        flat[c * 512 + c5[j * 3 + 0] * 64 + c5[j * 3 + 1] * 8 + c5[j * 3 + 2]] = fin[e];
    }
    __syncthreads();
    float s0 = 0.0f, s1 = 0.0f;
    for (int i = t; i < 1024; i += blockDim.x) {
        float f = flat[i];
        s0 += f * Wl[i * 2 + 0];
        s1 += f * Wl[i * 2 + 1];
    }
    part[t][0] = s0; part[t][1] = s1;
    __syncthreads();
    if (t == 0) {
        float h0 = bl[0], h1 = bl[1];
        for (int i = 0; i < (int)blockDim.x; ++i) { h0 += part[i][0]; h1 += part[i][1]; }
        h0 = (h0 > 0.0f) ? h0 : (expf(h0) - 1.0f);   // elu
        h1 = (h1 > 0.0f) ? h1 : (expf(h1) - 1.0f);
        float z = h0 * W3[0] + h1 * W3[1] + b3[0];
        out[0] = 1.0f / (1.0f + expf(-z));            // sigmoid
    }
}

// ---------------------------------------------------------------------------
// Dispatch over the 10 channel geometries used by the network
// ---------------------------------------------------------------------------
static void launch_conv(const __bf16* X, const __bf16* Wt, const int* nbr,
                        const __bf16* zrow, const float* bn, const __bf16* resid,
                        __bf16* Y, int K, int Ci, int Co, int Nin, int Nout, int mode,
                        hipStream_t stream) {
    const int waves = (Nout + 15) / 16;
    const dim3 g((waves + 7) / 8), b(256);   // 8 waves per 256-thread block
#define CONV_CASE(ci, co)                                                               \
    if (Ci == (ci) && Co == (co)) {                                                     \
        if (mode == 1)                                                                  \
            subm_conv_wmma<ci, co, 1><<<g, b, 0, stream>>>(X, Wt, nbr, zrow, bn,        \
                bn + (co), resid, Y, K, Nin, Nout);                                     \
        else                                                                            \
            subm_conv_wmma<ci, co, 2><<<g, b, 0, stream>>>(X, Wt, nbr, zrow, bn,        \
                bn + (co), resid, Y, K, Nin, Nout);                                     \
        return;                                                                         \
    }
    CONV_CASE(32, 32)  CONV_CASE(64, 64)   CONV_CASE(96, 96)
    CONV_CASE(128, 128) CONV_CASE(160, 160)
    CONV_CASE(32, 64)  CONV_CASE(64, 96)   CONV_CASE(96, 128)
    CONV_CASE(128, 160) CONV_CASE(160, 192)
#undef CONV_CASE
}

// ---------------------------------------------------------------------------
// Host orchestration
// ---------------------------------------------------------------------------
extern "C" void kernel_launch(void* const* d_in, const int* in_sizes, int n_in,
                              void* d_out, int out_size, void* d_ws, size_t ws_size,
                              hipStream_t stream) {
    (void)n_in; (void)out_size; (void)ws_size;

    const float* x_feat = (const float*)d_in[0];
    const float* W_init = (const float*)d_in[1];
    const float *Wa[5], *bna[5], *Wb[5], *bnb[5], *Wd[5], *bnd[5];
    for (int i = 0; i < 5; ++i) {
        Wa[i]  = (const float*)d_in[2 + 4 * i];
        bna[i] = (const float*)d_in[3 + 4 * i];
        Wb[i]  = (const float*)d_in[4 + 4 * i];
        bnb[i] = (const float*)d_in[5 + 4 * i];
        Wd[i]  = (const float*)d_in[22 + 2 * i];
        bnd[i] = (const float*)d_in[23 + 2 * i];
    }
    const float* W_fin  = (const float*)d_in[32];
    const float* W_lin  = (const float*)d_in[33];
    const float* b_lin  = (const float*)d_in[34];
    const float* W_lin3 = (const float*)d_in[35];
    const float* b_lin3 = (const float*)d_in[36];
    const int* nbr0 = (const int*)d_in[37];
    const int* nbrS[5]; for (int i = 0; i < 5; ++i) nbrS[i] = (const int*)d_in[38 + i];
    const int* nbr6 = (const int*)d_in[43];
    const int* nbrD[5]; for (int i = 0; i < 5; ++i) nbrD[i] = (const int*)d_in[44 + i];
    const int* coords5 = (const int*)d_in[49];

    const int Cs[6] = {32, 64, 96, 128, 160, 192};
    const int Kd[5] = {8, 8, 27, 8, 8};
    int Ns[6];
    Ns[0] = in_sizes[0];                          // x_feat rows
    for (int i = 0; i < 5; ++i) Ns[i + 1] = in_sizes[44 + i] / Kd[i];

    // ---- workspace bump allocator (256B aligned) ----
    char* base = (char*)d_ws;
    size_t off = 0;
    auto alloc = [&](size_t bytes) -> void* {
        void* p = base + off;
        off = (off + bytes + 255) & ~(size_t)255;
        return p;
    };
    auto r16 = [](int n) { return (n + 15) & ~15; };
    size_t maxE = 0;
    for (int i = 0; i < 6; ++i) {
        size_t e = (size_t)r16(Ns[i]) * Cs[i];    // row-padded for unconditional stores
        if (e > maxE) maxE = e;
    }
    __bf16* fA   = (__bf16*)alloc(maxE * sizeof(__bf16));
    __bf16* fB   = (__bf16*)alloc(maxE * sizeof(__bf16));
    __bf16* zrow = (__bf16*)alloc(192 * sizeof(__bf16));
    __bf16 *WaP[5], *WbP[5], *WdP[5];
    float  *bnA[5], *bnB[5], *bnD[5];             // [scale(C) | shift(C)]
    for (int i = 0; i < 5; ++i) {
        WaP[i] = (__bf16*)alloc((size_t)27 * Cs[i] * Cs[i] * sizeof(__bf16));
        WbP[i] = (__bf16*)alloc((size_t)27 * Cs[i] * Cs[i] * sizeof(__bf16));
        WdP[i] = (__bf16*)alloc((size_t)Kd[i] * Cs[i] * Cs[i + 1] * sizeof(__bf16));
        bnA[i] = (float*)alloc(2 * Cs[i] * sizeof(float));
        bnB[i] = (float*)alloc(2 * Cs[i] * sizeof(float));
        bnD[i] = (float*)alloc(2 * Cs[i + 1] * sizeof(float));
    }
    float* fin = (float*)alloc((size_t)Ns[5] * 2 * sizeof(float));

    // ---- prep: zero row, BN folds, weight packing ----
    zero_bf16_kernel<<<1, 256, 0, stream>>>(zrow, 192);
    for (int i = 0; i < 5; ++i) {
        int C = Cs[i], Co = Cs[i + 1];
        bn_prep_kernel<<<(C + 63) / 64, 64, 0, stream>>>(bna[i], bnA[i], bnA[i] + C, C);
        bn_prep_kernel<<<(C + 63) / 64, 64, 0, stream>>>(bnb[i], bnB[i], bnB[i] + C, C);
        bn_prep_kernel<<<(Co + 63) / 64, 64, 0, stream>>>(bnd[i], bnD[i], bnD[i] + Co, Co);
        int na = 27 * C * C;
        pack_w_kernel<<<(na + 255) / 256, 256, 0, stream>>>(Wa[i], WaP[i], 27, C, C);
        pack_w_kernel<<<(na + 255) / 256, 256, 0, stream>>>(Wb[i], WbP[i], 27, C, C);
        int nd = Kd[i] * C * Co;
        pack_w_kernel<<<(nd + 255) / 256, 256, 0, stream>>>(Wd[i], WdP[i], Kd[i], C, Co);
    }

    // ---- initial conv (Cin=1) ----
    conv0_kernel<<<(Ns[0] * 32 + 255) / 256, 256, 0, stream>>>(x_feat, W_init, nbr0, fA, Ns[0]);

    // ---- body: 5 levels of (2 residual blocks + downsample) ----
    __bf16* cur = fA;
    __bf16* alt = fB;
    for (int i = 0; i < 5; ++i) {
        const int C = Cs[i], N = Ns[i];
        for (int rep = 0; rep < 2; ++rep) {
            // h1 = leaky(bn(conv(x)))
            launch_conv(cur, WaP[i], nbrS[i], zrow, bnA[i], nullptr, alt,
                        27, C, C, N, N, 1, stream);
            // x = x + bn(conv(h1))   (in-place residual: per-wave ownership)
            launch_conv(alt, WbP[i], nbrS[i], zrow, bnB[i], cur, cur,
                        27, C, C, N, N, 2, stream);
        }
        // downsample: x' = leaky(bn(conv_stride2(x)))
        launch_conv(cur, WdP[i], nbrD[i], zrow, bnD[i], nullptr, alt,
                    Kd[i], C, Cs[i + 1], N, Ns[i + 1], 1, stream);
        __bf16* t = cur; cur = alt; alt = t;
    }

    // ---- final conv + head ----
    fin_conv_kernel<<<(Ns[5] * 2 + 255) / 256, 256, 0, stream>>>(cur, W_fin, nbr6, fin, Ns[5]);
    head_kernel<<<1, 256, 0, stream>>>(fin, coords5, Ns[5], W_lin, b_lin, W_lin3, b_lin3,
                                       (float*)d_out);
}
